// MLSTMCell_18442589569174
// MI455X (gfx1250) — matrix operand order repaired and checked by reading, not compiled, not run on hardware
//
#include <hip/hip_runtime.h>
#include <hip/hip_bf16.h>

// ---------------- problem constants ----------------
#define BB   2048
#define IN_  512
#define HH   512
#define KK   32
#define OUT_ 512
#define XW   1536          // IN + 2H (columns of packed X; gates use first 1024)

typedef __attribute__((ext_vector_type(16))) _Float16 v16h;
typedef __attribute__((ext_vector_type(8)))  _Float16 v8h;
typedef __attribute__((ext_vector_type(4)))  _Float16 v4h;
typedef __attribute__((ext_vector_type(8)))  float    v8f;

// ---------------- helpers ----------------
__device__ __forceinline__ float sigmoidf(float x) {
  return 1.0f / (1.0f + __expf(-x));
}

// A fragment: X row-major [*, XW]; lane holds row m = tm*16 + (lane&15).
// halves 0..7  -> K = k0 + hi*8 + {0..7}
// halves 8..15 -> K = k0 + 16 + hi*8 + {0..7}
__device__ __forceinline__ v16h load_a_frag(const _Float16* __restrict__ row,
                                            int k0, int hi) {
  const v8h c0 = *(const v8h*)(row + k0 + hi * 8);
  const v8h c1 = *(const v8h*)(row + k0 + 16 + hi * 8);
  v16h a;
#pragma unroll
  for (int i = 0; i < 8; ++i) { a[i] = c0[i]; a[i + 8] = c1[i]; }
  return a;
}

// B fragment: W row-major [N, Kdim]; lane holds row n = tn*16 + (lane&15),
// halves 0..7 -> K = k0 + hi*8 + {0..7}, halves 8..15 -> K = k0 + 16 + hi*8 + {0..7}.
__device__ __forceinline__ v16h load_b_frag(const _Float16* __restrict__ row,
                                            int k0, int hi) {
  const v8h c0 = *(const v8h*)(row + k0 + hi * 8);
  const v8h c1 = *(const v8h*)(row + k0 + 16 + hi * 8);
  v16h b;
#pragma unroll
  for (int i = 0; i < 8; ++i) { b[i] = c0[i]; b[i + 8] = c1[i]; }
  return b;
}

#define WMMA_F16(A, Bm, C) \
  __builtin_amdgcn_wmma_f32_16x16x32_f16(false, (A), false, (Bm), (short)0, (C), false, false)

// ---------------- prep kernels ----------------
__global__ void cvt_f32_to_f16(const float* __restrict__ src,
                               _Float16* __restrict__ dst, int n) {
  int i = blockIdx.x * blockDim.x + threadIdx.x;
  if (i < n) dst[i] = (_Float16)src[i];
}

// X16[b, 0:512]=sample, [512:1024]=hidden, [1024:1536]=d_0
__global__ void pack_x_kernel(const float* __restrict__ sample,
                              const float* __restrict__ hidden,
                              const float* __restrict__ d0,
                              _Float16* __restrict__ X) {
  int i = blockIdx.x * blockDim.x + threadIdx.x;
  const int n = BB * XW;
  if (i >= n) return;
  int b = i / XW, c = i - b * XW;
  float v;
  if (c < IN_)            v = sample[b * IN_ + c];
  else if (c < IN_ + HH)  v = hidden[b * HH + (c - IN_)];
  else                    v = d0[b * HH + (c - IN_ - HH)];
  X[i] = (_Float16)v;
}

// ---------------- fused 4-gate WMMA GEMM ----------------
// One wave computes one 16x16 tile of all four gates.
// i_gate/o_gate/c_tilde -> ws (f32), d_values -> d_out region (f32).
__global__ __launch_bounds__(256) void gemm_gates_kernel(
    const _Float16* __restrict__ X,      // [BB, XW] f16
    const _Float16* __restrict__ Wc16,   // [HH, 1024]
    const _Float16* __restrict__ Wi16,   // [HH, 1024]
    const _Float16* __restrict__ Wf16,   // [HH, 1536]
    const _Float16* __restrict__ Wo16,   // [HH, 1024]
    const float* __restrict__ bc, const float* __restrict__ bi,
    const float* __restrict__ bf, const float* __restrict__ bo,
    float* __restrict__ ig, float* __restrict__ og,
    float* __restrict__ ctl, float* __restrict__ dval) {
  const int wave = (blockIdx.x * blockDim.x + threadIdx.x) >> 5;
  const int lane = threadIdx.x & 31;
  const int tiles_n = HH / 16;                 // 32
  const int tn = wave % tiles_n;
  const int tm = wave / tiles_n;
  if (tm >= BB / 16) return;                   // wave-uniform guard
  const int lo = lane & 15, hi = lane >> 4;

  const _Float16* xrow  = X    + (size_t)(tm * 16 + lo) * XW;
  const _Float16* wcrow = Wc16 + (size_t)(tn * 16 + lo) * 1024;
  const _Float16* wirow = Wi16 + (size_t)(tn * 16 + lo) * 1024;
  const _Float16* wfrow = Wf16 + (size_t)(tn * 16 + lo) * 1536;
  const _Float16* worow = Wo16 + (size_t)(tn * 16 + lo) * 1024;

  v8f accC = {}, accI = {}, accF = {}, accO = {};
#pragma unroll 4
  for (int k0 = 0; k0 < 1024; k0 += 32) {
    v16h a = load_a_frag(xrow, k0, hi);
    accC = WMMA_F16(a, load_b_frag(wcrow, k0, hi), accC);
    accI = WMMA_F16(a, load_b_frag(wirow, k0, hi), accI);
    accF = WMMA_F16(a, load_b_frag(wfrow, k0, hi), accF);
    accO = WMMA_F16(a, load_b_frag(worow, k0, hi), accO);
  }
#pragma unroll 4
  for (int k0 = 1024; k0 < 1536; k0 += 32) {   // d_0 columns: Wf only
    v16h a = load_a_frag(xrow, k0, hi);
    accF = WMMA_F16(a, load_b_frag(wfrow, k0, hi), accF);
  }

  const int n = tn * 16 + lo;
  const float vbc = bc[n], vbi = bi[n], vbf = bf[n], vbo = bo[n];
#pragma unroll
  for (int v = 0; v < 8; ++v) {
    const int m = tm * 16 + v + 8 * hi;
    const size_t idx = (size_t)m * HH + n;
    ctl[idx]  = tanhf(accC[v] + vbc);
    ig[idx]   = sigmoidf(accI[v] + vbi);
    og[idx]   = sigmoidf(accO[v] + vbo);
    dval[idx] = 0.5f * sigmoidf(accF[v] + vbf);
  }
}

// ---------------- fused streaming kernel ----------------
// Per float4 column: cumprod fractional weights, weighted cell_tensor
// reduction, shifted h_c_1 copy, cell/hidden_new, f16 hidden copy.
__global__ __launch_bounds__(256) void fuse_kernel(
    const float* __restrict__ cell_tensor,  // [KK, BB*HH]
    const float* __restrict__ ig, const float* __restrict__ og,
    const float* __restrict__ ctl, const float* __restrict__ dval,
    float* __restrict__ out_hidden,         // [BB*HH]
    float* __restrict__ out_hc1,            // [KK, BB*HH]
    _Float16* __restrict__ h16) {           // [BB*HH] f16
  const size_t NV = (size_t)BB * HH / 4;
  size_t idx4 = (size_t)blockIdx.x * blockDim.x + threadIdx.x;
  if (idx4 >= NV) return;

  const float4* ct4  = (const float4*)cell_tensor;
  float4*       hc4  = (float4*)out_hc1;
  const float4 d = ((const float4*)dval)[idx4];

  float px = 1.0f, py = 1.0f, pz = 1.0f, pw = 1.0f;
  float fx = 0.0f, fy = 0.0f, fz = 0.0f, fw = 0.0f;
#pragma unroll
  for (int i = 0; i < KK; ++i) {
    const int j = KK - 1 - i;
    const float fi = (float)i;
    const float inv = 1.0f / (fi + 1.0f);
    px *= (fi - d.x) * inv;
    py *= (fi - d.y) * inv;
    pz *= (fi - d.z) * inv;
    pw *= (fi - d.w) * inv;
    const float4 c = ct4[(size_t)j * NV + idx4];
    fx -= c.x * px; fy -= c.y * py; fz -= c.z * pz; fw -= c.w * pw;
    if (j >= 1) hc4[(size_t)(j - 1) * NV + idx4] = c;   // shifted copy
  }

  const float4 iG = ((const float4*)ig)[idx4];
  const float4 oG = ((const float4*)og)[idx4];
  const float4 cT = ((const float4*)ctl)[idx4];
  float4 cell;
  cell.x = fx + cT.x * iG.x;
  cell.y = fy + cT.y * iG.y;
  cell.z = fz + cT.z * iG.z;
  cell.w = fw + cT.w * iG.w;
  hc4[(size_t)(KK - 1) * NV + idx4] = cell;

  float4 hn;
  hn.x = tanhf(cell.x) * oG.x;
  hn.y = tanhf(cell.y) * oG.y;
  hn.z = tanhf(cell.z) * oG.z;
  hn.w = tanhf(cell.w) * oG.w;
  ((float4*)out_hidden)[idx4] = hn;

  v4h hv = {(_Float16)hn.x, (_Float16)hn.y, (_Float16)hn.z, (_Float16)hn.w};
  *(v4h*)(h16 + idx4 * 4) = hv;
}

// ---------------- output WMMA GEMM ----------------
__global__ __launch_bounds__(256) void gemm_out_kernel(
    const _Float16* __restrict__ Hm,      // [BB, HH] f16
    const _Float16* __restrict__ W16,     // [OUT_, HH] f16
    const float* __restrict__ bout,
    float* __restrict__ out) {            // [BB, OUT_]
  const int wave = (blockIdx.x * blockDim.x + threadIdx.x) >> 5;
  const int lane = threadIdx.x & 31;
  const int tiles_n = OUT_ / 16;          // 32
  const int tn = wave % tiles_n;
  const int tm = wave / tiles_n;
  if (tm >= BB / 16) return;
  const int lo = lane & 15, hi = lane >> 4;

  const _Float16* xrow = Hm  + (size_t)(tm * 16 + lo) * HH;
  const _Float16* wrow = W16 + (size_t)(tn * 16 + lo) * HH;

  v8f acc = {};
#pragma unroll 4
  for (int k0 = 0; k0 < HH; k0 += 32) {
    v16h a = load_a_frag(xrow, k0, hi);
    acc = WMMA_F16(a, load_b_frag(wrow, k0, hi), acc);
  }

  const int n = tn * 16 + lo;
  const float vb = bout[n];
#pragma unroll
  for (int v = 0; v < 8; ++v) {
    const int m = tm * 16 + v + 8 * hi;
    out[(size_t)m * OUT_ + n] = acc[v] + vb;
  }
}

// ---------------- launcher ----------------
extern "C" void kernel_launch(void* const* d_in, const int* in_sizes, int n_in,
                              void* d_out, int out_size, void* d_ws, size_t ws_size,
                              hipStream_t stream) {
  const float* sample      = (const float*)d_in[0];
  const float* hidden      = (const float*)d_in[1];
  const float* cell_tensor = (const float*)d_in[2];
  const float* d_0         = (const float*)d_in[3];
  const float* Wc          = (const float*)d_in[4];
  const float* bc          = (const float*)d_in[5];
  const float* Wi          = (const float*)d_in[6];
  const float* bi          = (const float*)d_in[7];
  const float* Wf          = (const float*)d_in[8];
  const float* bf          = (const float*)d_in[9];
  const float* Wo          = (const float*)d_in[10];
  const float* bo          = (const float*)d_in[11];
  const float* Wout        = (const float*)d_in[12];
  const float* bout        = (const float*)d_in[13];

  // d_out = (output, hidden_new, h_c_1, d_values), all f32
  float* out_output = (float*)d_out;
  float* out_hidden = out_output + (size_t)BB * OUT_;
  float* out_hc1    = out_hidden + (size_t)BB * HH;
  float* out_dval   = out_hc1    + (size_t)KK * BB * HH;

  // workspace carve-out (256B aligned)
  char* ws = (char*)d_ws;
  size_t off = 0;
  auto carve = [&](size_t bytes) -> void* {
    void* p = ws + off;
    off = (off + bytes + 255) & ~(size_t)255;
    return p;
  };
  _Float16* X16    = (_Float16*)carve((size_t)BB * XW * 2);
  _Float16* Wc16   = (_Float16*)carve((size_t)HH * 1024 * 2);
  _Float16* Wi16   = (_Float16*)carve((size_t)HH * 1024 * 2);
  _Float16* Wo16   = (_Float16*)carve((size_t)HH * 1024 * 2);
  _Float16* Wf16   = (_Float16*)carve((size_t)HH * 1536 * 2);
  _Float16* Wout16 = (_Float16*)carve((size_t)OUT_ * HH * 2);
  float*    ig     = (float*)carve((size_t)BB * HH * 4);
  float*    og     = (float*)carve((size_t)BB * HH * 4);
  float*    ctl    = (float*)carve((size_t)BB * HH * 4);
  _Float16* h16    = (_Float16*)carve((size_t)BB * HH * 2);

  const int T = 256;
  // weight + input conversions to f16
  cvt_f32_to_f16<<<(HH * 1024 + T - 1) / T, T, 0, stream>>>(Wc, Wc16, HH * 1024);
  cvt_f32_to_f16<<<(HH * 1024 + T - 1) / T, T, 0, stream>>>(Wi, Wi16, HH * 1024);
  cvt_f32_to_f16<<<(HH * 1024 + T - 1) / T, T, 0, stream>>>(Wo, Wo16, HH * 1024);
  cvt_f32_to_f16<<<(HH * 1536 + T - 1) / T, T, 0, stream>>>(Wf, Wf16, HH * 1536);
  cvt_f32_to_f16<<<(OUT_ * HH + T - 1) / T, T, 0, stream>>>(Wout, Wout16, OUT_ * HH);
  pack_x_kernel<<<(BB * XW + T - 1) / T, T, 0, stream>>>(sample, hidden, d_0, X16);

  // fused four-gate WMMA GEMM: (BB/16)*(HH/16)=4096 waves, 8 waves/block
  gemm_gates_kernel<<<(BB / 16) * (HH / 16) / 8, T, 0, stream>>>(
      X16, Wc16, Wi16, Wf16, Wo16, bc, bi, bf, bo, ig, og, ctl, out_dval);

  // fused cumprod reduction + h_c_1 shift + hidden_new
  fuse_kernel<<<((size_t)BB * HH / 4 + T - 1) / T, T, 0, stream>>>(
      cell_tensor, ig, og, ctl, out_dval, out_hidden, out_hc1, h16);

  // output WMMA GEMM
  gemm_out_kernel<<<(BB / 16) * (OUT_ / 16) / 8, T, 0, stream>>>(
      h16, Wout16, bout, out_output);
}
